// VisionMamba_87857851006989
// MI455X (gfx1250) — compile-verified, matrix operand order checked by reading, not run on hardware
//
#include <hip/hip_runtime.h>
#include <hip/hip_bf16.h>

// ---------------- problem constants ----------------
#define B_    8
#define L_    2048
#define FIN   256     // DIM/4
#define D_    512     // SSM_RATIO*D_MODEL/4
#define N_    16
#define R_    64
#define C96   96      // R + 2N
#define BL_   (B_*L_) // 16384
#define NCH   16      // scan chunks
#define CHL   (L_/NCH)

typedef __bf16 bf16;
typedef __attribute__((ext_vector_type(16))) __bf16 v16bf;
typedef __attribute__((ext_vector_type(8)))  __bf16 v8bf;
typedef __attribute__((ext_vector_type(8)))  float  v8f;

// load a 16-element bf16 WMMA fragment as two aligned 128-bit vector loads
__device__ __forceinline__ v16bf load_frag(const bf16* p) {
  v8bf lo = *reinterpret_cast<const v8bf*>(p);        // K slice {base..base+7}
  v8bf hi = *reinterpret_cast<const v8bf*>(p + 16);   // K slice {base+16..base+23}
  return __builtin_shufflevector(lo, hi, 0,1,2,3,4,5,6,7,8,9,10,11,12,13,14,15);
}

// ---------------- fp32 -> bf16 conversion ----------------
__global__ void cvt_f32_bf16(const float* __restrict__ src, bf16* __restrict__ dst, int n) {
  for (int i = blockIdx.x * blockDim.x + threadIdx.x; i < n; i += gridDim.x * blockDim.x)
    dst[i] = (bf16)src[i];
}

// extract dts (first 64 cols of x_dbl, stride 96) to a dense bf16 (BL,64)
__global__ void cvt_dts(const float* __restrict__ xdbl, bf16* __restrict__ dts, int n) {
  for (int i = blockIdx.x * blockDim.x + threadIdx.x; i < n; i += gridDim.x * blockDim.x) {
    int row = i >> 6, r = i & 63;
    dts[i] = (bf16)xdbl[(size_t)row * C96 + r];
  }
}

// ---------------- generic WMMA bf16 GEMM, NT tiles in N per wave ----------------
// out[M,Ncols] = A[M,K] (row-major bf16) * W[Ncols,K]^T (row-major bf16), f32 accum.
// MODE: 0 = none, 1 = silu, 2 = softplus(x + bias[col])
// One wave computes a 16 x (16*NT) strip: A fragment loaded once per K-step,
// reused for NT WMMAs; B-fragment t+1 loads overlap WMMA t in the matrix pipe.
template<int MODE, int NT>
__global__ void gemm_wmma_bf16(const bf16* __restrict__ A, const bf16* __restrict__ W,
                               float* __restrict__ out, const float* __restrict__ bias,
                               int K, int Ncols, int groupsN)
{
  const int wave   = threadIdx.x >> 5;
  const int lane   = threadIdx.x & 31;
  const int tile   = blockIdx.x * (blockDim.x >> 5) + wave;
  const int tg     = tile % groupsN;          // group of NT n-tiles
  const int tm     = tile / groupsN;
  const int lane16 = lane & 15;
  const int half   = lane >> 4;

  // A operand: lane 0-15 row m=lane16 holds K {0..7,16..23}; lanes 16-31 hold {8..15,24..31}
  const bf16* arow = A + (size_t)(tm * 16 + lane16) * K + half * 8;
  // B operand fed as B^T rows (= rows of the weight matrix), same per-lane K slicing
  const bf16* wrow[NT];
#pragma unroll
  for (int t = 0; t < NT; ++t)
    wrow[t] = W + (size_t)((tg * NT + t) * 16 + lane16) * K + half * 8;

  v8f acc[NT];
#pragma unroll
  for (int t = 0; t < NT; ++t) acc[t] = (v8f){};

  for (int k0 = 0; k0 < K; k0 += 32) {
    v16bf av = load_frag(arow + k0);
    if (k0 + 32 < K) {                        // prefetch next K block -> global_prefetch_b8
      __builtin_prefetch(arow + k0 + 32, 0, 0);
#pragma unroll
      for (int t = 0; t < NT; ++t) __builtin_prefetch(wrow[t] + k0 + 32, 0, 0);
    }
#pragma unroll
    for (int t = 0; t < NT; ++t) {
      v16bf bv = load_frag(wrow[t] + k0);
      acc[t] = __builtin_amdgcn_wmma_f32_16x16x32_bf16(false, av, false, bv,
                                                       (short)0, acc[t], false, false);
    }
  }

#pragma unroll
  for (int t = 0; t < NT; ++t) {
    const int ncol = (tg * NT + t) * 16 + lane16;
    const float bs = (MODE == 2) ? bias[ncol] : 0.f;
#pragma unroll
    for (int i = 0; i < 8; ++i) {
      int m = tm * 16 + i + 8 * half;         // C/D layout: VGPR i -> M=i (lanes 0-15), M=i+8 (16-31)
      float v = acc[t][i];
      if (MODE == 1) v = v * (1.f / (1.f + __expf(-v)));
      if (MODE == 2) { v += bs; v = (v > 20.f) ? v : log1pf(__expf(v)); }
      out[(size_t)m * Ncols + ncol] = v;
    }
  }
}

// ---------------- selective scan, chunked ----------------
// thread -> (b, chunk, d); lanes are consecutive d (coalesced delta/u, uniform Bs/Cs)
__global__ void scan_phaseA(const float* __restrict__ delta, const float* __restrict__ xs,
                            const float* __restrict__ xdbl,  const float* __restrict__ A_logs,
                            float* __restrict__ cA, float* __restrict__ cP)
{
  int tid   = blockIdx.x * blockDim.x + threadIdx.x;
  int d     = tid & (D_ - 1);
  int rem   = tid >> 9;
  int chunk = rem & (NCH - 1);
  int b     = rem >> 4;

  float negA[N_], h[N_], ap[N_];
#pragma unroll
  for (int n = 0; n < N_; ++n) {
    negA[n] = -__expf(A_logs[d * N_ + n]);
    h[n] = 0.f; ap[n] = 1.f;
  }
  int l0 = chunk * CHL;
  for (int l = l0; l < l0 + CHL; ++l) {
    size_t row = (size_t)b * L_ + l;
    float dl = delta[row * D_ + d];
    float ul = xs[row * D_ + d];
    float du = dl * ul;
    const float* bsp = xdbl + row * C96 + R_;      // Bs[b,:,l] (uniform across wave)
#pragma unroll
    for (int n = 0; n < N_; ++n) {
      float dA = __expf(dl * negA[n]);
      h[n]  = dA * h[n] + du * bsp[n];
      ap[n] = ap[n] * dA;
    }
  }
  size_t cix = (((size_t)b * NCH + chunk) * D_ + d) * N_;
#pragma unroll
  for (int n = 0; n < N_; ++n) { cA[cix + n] = ap[n]; cP[cix + n] = h[n]; }
}

// sequential combine over the 16 chunks: h0_{c+1} = A_c * h0_c + P_c
__global__ void scan_combine(const float* __restrict__ cA, const float* __restrict__ cP,
                             float* __restrict__ h0buf)
{
  int tid = blockIdx.x * blockDim.x + threadIdx.x;   // 0 .. B*D-1
  int d = tid & (D_ - 1);
  int b = tid >> 9;
  float h[N_];
#pragma unroll
  for (int n = 0; n < N_; ++n) h[n] = 0.f;
  for (int c = 0; c < NCH; ++c) {
    size_t cix = (((size_t)b * NCH + c) * D_ + d) * N_;
#pragma unroll
    for (int n = 0; n < N_; ++n) {
      h0buf[cix + n] = h[n];
      h[n] = cA[cix + n] * h[n] + cP[cix + n];
    }
  }
}

__global__ void scan_phaseC(const float* __restrict__ delta, const float* __restrict__ xs,
                            const float* __restrict__ xdbl,  const float* __restrict__ A_logs,
                            const float* __restrict__ Ds,    const float* __restrict__ h0buf,
                            float* __restrict__ y)
{
  int tid   = blockIdx.x * blockDim.x + threadIdx.x;
  int d     = tid & (D_ - 1);
  int rem   = tid >> 9;
  int chunk = rem & (NCH - 1);
  int b     = rem >> 4;

  float negA[N_], h[N_];
  size_t cix = (((size_t)b * NCH + chunk) * D_ + d) * N_;
#pragma unroll
  for (int n = 0; n < N_; ++n) {
    negA[n] = -__expf(A_logs[d * N_ + n]);
    h[n] = h0buf[cix + n];
  }
  float Dsv = Ds[d];
  int l0 = chunk * CHL;
  for (int l = l0; l < l0 + CHL; ++l) {
    size_t row = (size_t)b * L_ + l;
    float dl = delta[row * D_ + d];
    float ul = xs[row * D_ + d];
    float du = dl * ul;
    const float* bsp = xdbl + row * C96 + R_;
    const float* csp = xdbl + row * C96 + R_ + N_;
    float acc = 0.f;
#pragma unroll
    for (int n = 0; n < N_; ++n) {
      float dA = __expf(dl * negA[n]);
      h[n] = dA * h[n] + du * bsp[n];
      acc += h[n] * csp[n];
    }
    y[row * D_ + d] = acc + ul * Dsv;
  }
}

// ---------------- LayerNorm(D) + gate, one wave per (b,l) row ----------------
__global__ void ln_gate(const float* __restrict__ y, const float* __restrict__ z,
                        const float* __restrict__ g, const float* __restrict__ bta,
                        bf16* __restrict__ gb)
{
  int wid  = (blockIdx.x * blockDim.x + threadIdx.x) >> 5;   // row
  int lane = threadIdx.x & 31;
  const float* yr = y + (size_t)wid * D_;
  float v[D_ / 32], s = 0.f, s2 = 0.f;
#pragma unroll
  for (int j = 0; j < D_ / 32; ++j) {
    v[j] = yr[lane + 32 * j];
    s += v[j]; s2 += v[j] * v[j];
  }
#pragma unroll
  for (int off = 16; off > 0; off >>= 1) {
    s  += __shfl_xor(s,  off, 32);
    s2 += __shfl_xor(s2, off, 32);
  }
  float mu = s * (1.f / D_);
  float var = s2 * (1.f / D_) - mu * mu;
  float rs = rsqrtf(var + 1e-5f);
#pragma unroll
  for (int j = 0; j < D_ / 32; ++j) {
    int d = lane + 32 * j;
    float yn = (v[j] - mu) * rs * g[d] + bta[d];
    gb[(size_t)wid * D_ + d] = (bf16)(yn * z[(size_t)wid * D_ + d]);
  }
}

// ---------------- launcher ----------------
extern "C" void kernel_launch(void* const* d_in, const int* in_sizes, int n_in,
                              void* d_out, int out_size, void* d_ws, size_t ws_size,
                              hipStream_t stream) {
  const float* x         = (const float*)d_in[0];   // (B,L,FIN)
  const float* xs        = (const float*)d_in[1];   // (B,L,D)
  const float* in_proj_w = (const float*)d_in[2];   // (2D,FIN)
  const float* x_proj_w  = (const float*)d_in[3];   // (96,D)
  const float* dt_w      = (const float*)d_in[4];   // (D,R)
  const float* dt_b      = (const float*)d_in[5];   // (D,)
  const float* A_logs    = (const float*)d_in[6];   // (D,N)
  const float* Ds        = (const float*)d_in[7];   // (D,)
  const float* ln_g      = (const float*)d_in[8];
  const float* ln_b      = (const float*)d_in[9];
  const float* out_proj_w= (const float*)d_in[10];  // (FIN,D)
  float* out = (float*)d_out;

  char* w = (char*)d_ws;
  size_t off = 0;
  auto alloc = [&](size_t bytes) -> void* {
    off = (off + 255) & ~(size_t)255;
    void* p = w + off; off += bytes; return p;
  };
  bf16* xb    = (bf16*)alloc((size_t)BL_ * FIN * 2);
  bf16* xsb   = (bf16*)alloc((size_t)BL_ * D_  * 2);
  bf16* wz    = (bf16*)alloc((size_t)D_  * FIN * 2);
  bf16* xpw   = (bf16*)alloc((size_t)C96 * D_  * 2);
  bf16* dtw   = (bf16*)alloc((size_t)D_  * R_  * 2);
  bf16* opw   = (bf16*)alloc((size_t)FIN * D_  * 2);
  float* zbuf = (float*)alloc((size_t)BL_ * D_  * 4);
  float* xdbl = (float*)alloc((size_t)BL_ * C96 * 4);
  bf16* dtsb  = (bf16*)alloc((size_t)BL_ * R_  * 2);
  float* delta= (float*)alloc((size_t)BL_ * D_  * 4);
  float* ybuf = (float*)alloc((size_t)BL_ * D_  * 4);
  bf16* gb    = (bf16*)alloc((size_t)BL_ * D_  * 2);
  float* cA   = (float*)alloc((size_t)B_ * NCH * D_ * N_ * 4);
  float* cP   = (float*)alloc((size_t)B_ * NCH * D_ * N_ * 4);
  float* h0b  = (float*)alloc((size_t)B_ * NCH * D_ * N_ * 4);

  // 1. bf16 conversions (only rows D..2D of in_proj_w are ever used -> silu branch)
  cvt_f32_bf16<<<4096, 256, 0, stream>>>(x,  xb,  BL_ * FIN);
  cvt_f32_bf16<<<4096, 256, 0, stream>>>(xs, xsb, BL_ * D_);
  cvt_f32_bf16<<<512,  256, 0, stream>>>(in_proj_w + (size_t)D_ * FIN, wz, D_ * FIN);
  cvt_f32_bf16<<<192,  256, 0, stream>>>(x_proj_w,  xpw, C96 * D_);
  cvt_f32_bf16<<<128,  256, 0, stream>>>(dt_w,      dtw, D_ * R_);
  cvt_f32_bf16<<<512,  256, 0, stream>>>(out_proj_w, opw, FIN * D_);

  // 2. z = silu(x @ Wz^T)          M=16384 N=512 K=256  NT=4 -> 8192 waves
  gemm_wmma_bf16<1, 4><<<1024, 256, 0, stream>>>(xb, wz, zbuf, nullptr, FIN, D_, D_ / 64);
  // 3. x_dbl = xs @ x_proj_w^T     M=16384 N=96  K=512  NT=2 -> 3072 waves
  gemm_wmma_bf16<0, 2><<<384, 256, 0, stream>>>(xsb, xpw, xdbl, nullptr, D_, C96, C96 / 32);
  // 4. dts -> bf16 dense
  cvt_dts<<<4096, 256, 0, stream>>>(xdbl, dtsb, BL_ * R_);
  // 5. delta = softplus(dts @ dt_w^T + dt_b)  M=16384 N=512 K=64  NT=4
  gemm_wmma_bf16<2, 4><<<1024, 256, 0, stream>>>(dtsb, dtw, delta, dt_b, R_, D_, D_ / 64);

  // 6. chunked selective scan (16x parallelism over L)
  scan_phaseA <<<(B_ * NCH * D_) / 256, 256, 0, stream>>>(delta, xs, xdbl, A_logs, cA, cP);
  scan_combine<<<(B_ * D_) / 256,       256, 0, stream>>>(cA, cP, h0b);
  scan_phaseC <<<(B_ * NCH * D_) / 256, 256, 0, stream>>>(delta, xs, xdbl, A_logs, Ds, h0b, ybuf);

  // 7. LayerNorm + gate -> bf16
  ln_gate<<<BL_ / 8, 256, 0, stream>>>(ybuf, zbuf, ln_g, ln_b, gb);

  // 8. out = g @ out_proj_w^T      M=16384 N=256 K=512  NT=4
  gemm_wmma_bf16<0, 4><<<512, 256, 0, stream>>>(gb, opw, out, nullptr, D_, FIN, FIN / 64);
}
